// VectorQuantize_27084063768819
// MI455X (gfx1250) — matrix operand order repaired and checked by reading, not compiled, not run on hardware
//
#include <hip/hip_runtime.h>

#define NTOK   65536
#define CB_K   512
#define CB_D   64
#define ROWPAD 72   // padded bf16 row stride in LDS (144B -> bank-spread, 16B aligned)

typedef __attribute__((ext_vector_type(16))) __bf16 v16bf;
typedef __attribute__((ext_vector_type(8)))  __bf16 v8bf;
typedef __attribute__((ext_vector_type(8)))  float  v8f;

__device__ __forceinline__ v16bf load_b16x16(const __bf16* p) {
  // two 16B LDS loads (K offsets 0..7 and 16..23), combined into one 32B fragment
  v8bf lo = *(const v8bf*)p;
  v8bf hi = *(const v8bf*)(p + 16);
  return __builtin_shufflevector(lo, hi, 0,1,2,3,4,5,6,7,8,9,10,11,12,13,14,15);
}

__global__ __launch_bounds__(256)
void vq_wmma_kernel(const float* __restrict__ cb,
                    const float* __restrict__ emb,
                    float* __restrict__ out) {
  extern __shared__ char smem[];
  __bf16* sBhi = (__bf16*)smem;                       // [512][ROWPAD]
  __bf16* sBlo = sBhi + CB_K * ROWPAD;                // [512][ROWPAD]
  float*  sCsq = (float*)(sBlo + CB_K * ROWPAD);      // [512]
  int*    sIdx = (int*)(sCsq + CB_K);                 // [256] per-block winner indices

  const int tid = threadIdx.x;

  // ---- cooperative setup: codebook -> bf16 hi/lo in LDS + per-code |c|^2 ----
  for (int i = tid; i < CB_K * CB_D; i += 256) {
    int code = i >> 6, k = i & 63;
    float x = cb[i];
    __bf16 h = (__bf16)x;                 // RNE high part
    __bf16 l = (__bf16)(x - (float)h);    // residual
    sBhi[code * ROWPAD + k] = h;
    sBlo[code * ROWPAD + k] = l;
  }
  for (int c = tid; c < CB_K; c += 256) {
    float acc = 0.f;
    for (int k = 0; k < CB_D; ++k) { float x = cb[c * 64 + k]; acc = fmaf(x, x, acc); }
    sCsq[c] = acc;
  }
  __syncthreads();

  const int wave = tid >> 5;          // 8 waves per block (wave32)
  const int lane = tid & 31;
  const int sub  = lane >> 4;         // which 16-lane half
  const int n    = lane & 15;
  const int token_base = blockIdx.x * 256 + wave * 32;   // 32 tokens per wave

  // ---- build A fragments (2 tokensets x 16 tokens x 64, bf16 hi/lo), WMMA A layout ----
  // element e at lane l: M = l&15, K = e + 8*((e>=8) + (l>=16)) + 32*kstep
  v16bf ah[2][2], al[2][2];           // [tokenset][kstep]
  #pragma unroll
  for (int ts = 0; ts < 2; ++ts) {
    const float4* e4 = (const float4*)(emb + (size_t)(token_base + ts * 16 + n) * 64);
    #pragma unroll
    for (int ks = 0; ks < 2; ++ks) {
      float4 c0a = e4[8 * ks + 2 * sub + 0];      // K = 32ks + 8*sub + 0..7      -> e = 0..7
      float4 c0b = e4[8 * ks + 2 * sub + 1];
      float4 c1a = e4[8 * ks + 4 + 2 * sub + 0];  // K = 32ks + 16 + 8*sub + 0..7 -> e = 8..15
      float4 c1b = e4[8 * ks + 4 + 2 * sub + 1];
      float xs[16] = {c0a.x,c0a.y,c0a.z,c0a.w, c0b.x,c0b.y,c0b.z,c0b.w,
                      c1a.x,c1a.y,c1a.z,c1a.w, c1b.x,c1b.y,c1b.z,c1b.w};
      #pragma unroll
      for (int e = 0; e < 16; ++e) {
        float x = xs[e];
        __bf16 h = (__bf16)x;
        ah[ts][ks][e] = h;
        al[ts][ks][e] = (__bf16)(x - (float)h);
      }
    }
  }

  // ---- running argmin over 32 code tiles ----
  float mval[2][8]; int midx[2][8];
  #pragma unroll
  for (int ts = 0; ts < 2; ++ts)
    #pragma unroll
    for (int v = 0; v < 8; ++v) { mval[ts][v] = 3.4e38f; midx[ts][v] = 0; }

  #pragma unroll 2
  for (int t = 0; t < 32; ++t) {
    // B layout: lane -> column (code), K = 8*sub + {0..7, 16..23} per code row (matches A)
    const __bf16* pbh = sBhi + (t * 16 + n) * ROWPAD + 8 * sub;
    const __bf16* pbl = sBlo + (t * 16 + n) * ROWPAD + 8 * sub;
    v16bf bh0 = load_b16x16(pbh);         // K = 0..31  slice, hi part
    v16bf bh1 = load_b16x16(pbh + 32);    // K = 32..63 slice, hi part
    v16bf bl0 = load_b16x16(pbl);         // lo part
    v16bf bl1 = load_b16x16(pbl + 32);

    // four independent accumulator chains (2 tokensets x 2 ksteps), 3 WMMAs each:
    // split product hi*hi + lo*hi + hi*lo
    v8f a00 = {0.f,0.f,0.f,0.f,0.f,0.f,0.f,0.f};
    v8f a01 = {0.f,0.f,0.f,0.f,0.f,0.f,0.f,0.f};
    v8f a10 = {0.f,0.f,0.f,0.f,0.f,0.f,0.f,0.f};
    v8f a11 = {0.f,0.f,0.f,0.f,0.f,0.f,0.f,0.f};
    a00 = __builtin_amdgcn_wmma_f32_16x16x32_bf16(false, ah[0][0], false, bh0, (short)0, a00, false, false);
    a10 = __builtin_amdgcn_wmma_f32_16x16x32_bf16(false, ah[1][0], false, bh0, (short)0, a10, false, false);
    a01 = __builtin_amdgcn_wmma_f32_16x16x32_bf16(false, ah[0][1], false, bh1, (short)0, a01, false, false);
    a11 = __builtin_amdgcn_wmma_f32_16x16x32_bf16(false, ah[1][1], false, bh1, (short)0, a11, false, false);
    a00 = __builtin_amdgcn_wmma_f32_16x16x32_bf16(false, al[0][0], false, bh0, (short)0, a00, false, false);
    a10 = __builtin_amdgcn_wmma_f32_16x16x32_bf16(false, al[1][0], false, bh0, (short)0, a10, false, false);
    a01 = __builtin_amdgcn_wmma_f32_16x16x32_bf16(false, al[0][1], false, bh1, (short)0, a01, false, false);
    a11 = __builtin_amdgcn_wmma_f32_16x16x32_bf16(false, al[1][1], false, bh1, (short)0, a11, false, false);
    a00 = __builtin_amdgcn_wmma_f32_16x16x32_bf16(false, ah[0][0], false, bl0, (short)0, a00, false, false);
    a10 = __builtin_amdgcn_wmma_f32_16x16x32_bf16(false, ah[1][0], false, bl0, (short)0, a10, false, false);
    a01 = __builtin_amdgcn_wmma_f32_16x16x32_bf16(false, ah[0][1], false, bl1, (short)0, a01, false, false);
    a11 = __builtin_amdgcn_wmma_f32_16x16x32_bf16(false, ah[1][1], false, bl1, (short)0, a11, false, false);

    const float csq = sCsq[t * 16 + n];
    const int  cidx = t * 16 + n;
    #pragma unroll
    for (int v = 0; v < 8; ++v) {
      float d0 = fmaf(-2.0f, a00[v] + a01[v], csq);   // e_sq constant per row: dropped
      float d1 = fmaf(-2.0f, a10[v] + a11[v], csq);
      bool b0 = d0 < mval[0][v];
      bool b1 = d1 < mval[1][v];
      mval[0][v] = b0 ? d0   : mval[0][v];
      midx[0][v] = b0 ? cidx : midx[0][v];
      mval[1][v] = b1 ? d1   : mval[1][v];
      midx[1][v] = b1 ? cidx : midx[1][v];
    }
  }

  // ---- cross-lane argmin within each 16-lane half (tie -> lower index) ----
  #pragma unroll
  for (int mask = 1; mask < 16; mask <<= 1) {
    #pragma unroll
    for (int ts = 0; ts < 2; ++ts)
      #pragma unroll
      for (int v = 0; v < 8; ++v) {
        float ov = __shfl_xor(mval[ts][v], mask, 32);
        int   oi = __shfl_xor(midx[ts][v], mask, 32);
        bool take = (ov < mval[ts][v]) || (ov == mval[ts][v] && oi < midx[ts][v]);
        mval[ts][v] = take ? ov : mval[ts][v];
        midx[ts][v] = take ? oi : midx[ts][v];
      }
  }
  if (n == 0) {
    #pragma unroll
    for (int ts = 0; ts < 2; ++ts)
      #pragma unroll
      for (int v = 0; v < 8; ++v)
        sIdx[wave * 32 + ts * 16 + sub * 8 + v] = midx[ts][v];  // token m = ts*16 + v + 8*sub
  }
  __syncthreads();

  // ---- gather winning codebook rows (L2-hot) and stream outputs ----
  float* outQ  = out;
  float* outQ2 = out + (size_t)NTOK * 64;
  float* outI  = out + (size_t)NTOK * 128;
  #pragma unroll
  for (int ts = 0; ts < 2; ++ts) {
    const int tk  = token_base + ts * 16 + n;
    const int idx = sIdx[wave * 32 + ts * 16 + n];
    const float4* src = (const float4*)(cb + (size_t)idx * 64) + sub * 8;
    float4* dq  = (float4*)(outQ  + (size_t)tk * 64) + sub * 8;
    float4* dq2 = (float4*)(outQ2 + (size_t)tk * 64) + sub * 8;
    #pragma unroll
    for (int c = 0; c < 8; ++c) { float4 vv = src[c]; dq[c] = vv; dq2[c] = vv; }
  }
  outI[token_base + lane] = (float)sIdx[wave * 32 + lane];
}

extern "C" void kernel_launch(void* const* d_in, const int* in_sizes, int n_in,
                              void* d_out, int out_size, void* d_ws, size_t ws_size,
                              hipStream_t stream) {
  const float* cb  = (const float*)d_in[0];   // (512, 64) f32
  const float* emb = (const float*)d_in[1];   // (65536, 1, 64) f32
  float* out = (float*)d_out;                 // 65536*64 + 65536*64 + 65536 floats
  // LDS: 2 * 512*72 bf16 + 512 f32 + 256 i32 = 150,528 B (fits 320KB WGP LDS)
  const size_t shmem = (size_t)CB_K * ROWPAD * 2u * 2u + CB_K * 4u + 256u * 4u;
  dim3 grid(NTOK / 256), block(256);
  vq_wmma_kernel<<<grid, block, shmem, stream>>>(cb, emb, out);
}
